// CSRA_18923625906930
// MI455X (gfx1250) — matrix operand order repaired and checked by reading, not compiled, not run on hardware
//
#include <hip/hip_runtime.h>
#include <hip/hip_bf16.h>

typedef _Float16 h16;
typedef __attribute__((ext_vector_type(16))) _Float16 v16h;
typedef __attribute__((ext_vector_type(8)))  _Float16 v8h;
typedef __attribute__((ext_vector_type(8)))  float    v8f;

#define NB  2      // batch
#define NC  128    // C
#define NH  64
#define NW  64
#define NL  4096   // H*W
#define NDI 256
#define NST 16     // state N
#define NR  8
#define NK  4
#define MREP 4     // M-tiles per wave (B-fragment register reuse)

// ---------------- WMMA NT-GEMM core ----------------
// A: [M x K] row-major (K contiguous), B: [N x K] row-major (K contiguous).
// 16-bit A 16x32 fragment (ISA 7.12.2): lane row = lane%16; halfs i=0..7 -> K = (lane/16)*8 + i,
// i=8..15 -> K = 16 + (lane/16)*8 + (i-8).  B 32x16: lane col = lane%16; same K mapping as A.
static __device__ __forceinline__ v16h frag_a(const h16* rowPtr, int lane, int k0) {
  const h16* p = rowPtr + k0 + ((lane >> 4) << 3);
  v8h a0 = *reinterpret_cast<const v8h*>(p);
  v8h a1 = *reinterpret_cast<const v8h*>(p + 16);
  v16h f;
#pragma unroll
  for (int i = 0; i < 8; ++i) { f[i] = a0[i]; f[i + 8] = a1[i]; }
  return f;
}
static __device__ __forceinline__ v16h frag_b(const h16* rowPtr, int lane, int k0) {
  const h16* p = rowPtr + k0 + ((lane >> 4) << 3);
  v8h b0 = *reinterpret_cast<const v8h*>(p);
  v8h b1 = *reinterpret_cast<const v8h*>(p + 16);
  v16h f;
#pragma unroll
  for (int i = 0; i < 8; ++i) { f[i] = b0[i]; f[i + 8] = b1[i]; }
  return f;
}
// One wave computes an (MREP*16) x 16 super-tile: B fragments loaded once into
// registers, reused across MREP A-tiles -> 4x WMMA work per weight fetch.
template <int KD, int MR, typename RowF>
static __device__ __forceinline__ void wmma_super(RowF rowp, const h16* brow, int lane, v8f* acc) {
  v16h bf[KD / 32];
#pragma unroll
  for (int kk = 0; kk < KD / 32; ++kk) bf[kk] = frag_b(brow, lane, kk * 32);
#pragma unroll
  for (int mt = 0; mt < MR; ++mt) {
    const h16* arow = rowp(mt);
    v8f a = {};
#pragma unroll
    for (int kk = 0; kk < KD / 32; ++kk) {
      v16h fa = frag_a(arow, lane, kk * 32);
      a = __builtin_amdgcn_wmma_f32_16x16x32_f16(false, fa, false, bf[kk], (short)0, a, false, false);
    }
    acc[mt] = a;
  }
}
// C/D layout: VGPR r holds row M = tileM + r + 8*(lane/16), col N = tileN + lane%16.

static __device__ __forceinline__ float dsigmoid(float x) { return 1.0f / (1.0f + __expf(-x)); }
static __device__ __forceinline__ float dgelu(float x) { return 0.5f * x * (1.0f + erff(x * 0.70710678118654752f)); }
static __device__ __forceinline__ float dsoftplus(float x) { return (x > 20.0f) ? x : log1pf(__expf(x)); }
// scan index -> spatial index for the 4 cross-scan directions
static __device__ __forceinline__ int scan_map(int k, int l) {
  int l2 = (k & 2) ? (NL - 1 - l) : l;
  return (k & 1) ? (((l2 & 63) << 6) | (l2 >> 6)) : l2;
}

// ---------------- small prep kernels ----------------
__global__ void k_cvt_f16(const float* __restrict__ s, h16* __restrict__ d, int n) {
  int i = blockIdx.x * blockDim.x + threadIdx.x;
  if (i < n) d[i] = static_cast<h16>(s[i]);
}
// x_proj_w (4,40,256) -> padded (4,48,256), rows 40..47 = 0
__global__ void k_pack_xproj(const float* __restrict__ s, h16* __restrict__ d) {
  int i = blockIdx.x * blockDim.x + threadIdx.x;
  if (i >= NK * 48 * NDI) return;
  int dd = i & 255; int c = (i >> 8) % 48; int k = i / (48 * NDI);
  d[i] = (c < 40) ? static_cast<h16>(s[(k * 40 + c) * NDI + dd]) : static_cast<h16>(0.0f);
}
// dt_proj_w (4,256,8) -> padded (4,256,32), r 8..31 = 0 (kills B/C channels in contraction)
__global__ void k_pack_dt(const float* __restrict__ s, h16* __restrict__ d) {
  int i = blockIdx.x * blockDim.x + threadIdx.x;
  if (i >= NK * NDI * 32) return;
  int r = i & 31; int dd = (i >> 5) & 255; int k = i >> 13;
  d[i] = (r < 8) ? static_cast<h16>(s[(k * NDI + dd) * NR + r]) : static_cast<h16>(0.0f);
}
// BN1 + transpose (B,C,H,W) -> (B*L, C) f16
__global__ void k_bn1_pack(const float* __restrict__ feat, const float* __restrict__ g,
                           const float* __restrict__ bt, const float* __restrict__ mn,
                           const float* __restrict__ vr, h16* __restrict__ A0) {
  int i = blockIdx.x * blockDim.x + threadIdx.x;
  if (i >= NB * NC * NL) return;
  int p = i & 4095; int c = (i >> 12) & 127; int b = i >> 19;
  float v = (feat[i] - mn[c]) * rsqrtf(vr[c] + 1e-5f) * g[c] + bt[c];
  A0[((size_t)b * NL + p) * NC + c] = static_cast<h16>(v);
}

// ---------------- GEMM 1: pre_w + BN + GELU ----------------
__global__ void k_gemm_pre(const h16* __restrict__ A, const h16* __restrict__ Bw,
                           const float* __restrict__ g, const float* __restrict__ bt,
                           const float* __restrict__ mn, const float* __restrict__ vr,
                           h16* __restrict__ Afeat) {
  int lane = threadIdx.x & 31;
  int tile = blockIdx.x * 8 + (threadIdx.x >> 5);
  const int NT = NC / 16, SM = NB * NL / (16 * MREP);
  if (tile >= SM * NT) return;
  int tm = (tile / NT) * (16 * MREP), tn = (tile % NT) * 16;
  v8f acc[MREP];
  wmma_super<NC, MREP>(
      [&](int mt) { return A + (size_t)(tm + mt * 16 + (lane & 15)) * NC; },
      Bw + (size_t)(tn + (lane & 15)) * NC, lane, acc);
  int n = tn + (lane & 15);
  float gg = g[n], bb = bt[n], mm = mn[n], rs = rsqrtf(vr[n] + 1e-5f);
#pragma unroll
  for (int mt = 0; mt < MREP; ++mt)
#pragma unroll
    for (int r = 0; r < 8; ++r) {
      int m = tm + mt * 16 + r + ((lane >> 4) << 3);
      float v = dgelu((acc[mt][r] - mm) * rs * gg + bb);
      Afeat[(size_t)m * NC + n] = static_cast<h16>(v);
    }
}

// ---------------- GEMM 2: in_proj -> x (to conv) and z ----------------
__global__ void k_gemm_inproj(const h16* __restrict__ A, const h16* __restrict__ Bw,
                              float* __restrict__ xconv, float* __restrict__ zbuf) {
  int lane = threadIdx.x & 31;
  int tile = blockIdx.x * 8 + (threadIdx.x >> 5);
  const int NT = 512 / 16, SM = NB * NL / (16 * MREP);
  if (tile >= SM * NT) return;
  int tm = (tile / NT) * (16 * MREP), tn = (tile % NT) * 16;
  v8f acc[MREP];
  wmma_super<NC, MREP>(
      [&](int mt) { return A + (size_t)(tm + mt * 16 + (lane & 15)) * NC; },
      Bw + (size_t)(tn + (lane & 15)) * NC, lane, acc);
  int n = tn + (lane & 15);
#pragma unroll
  for (int mt = 0; mt < MREP; ++mt)
#pragma unroll
    for (int r = 0; r < 8; ++r) {
      int m = tm + mt * 16 + r + ((lane >> 4) << 3);
      int b = m >> 12, p = m & 4095;
      if (n < NDI) xconv[((size_t)b * NDI + n) * NL + p] = acc[mt][r];
      else         zbuf[(size_t)m * NDI + (n - NDI)]     = acc[mt][r];
    }
}

// ---------------- depthwise 3x3 + bias + SiLU ----------------
__global__ void k_dwconv(const float* __restrict__ xin, const float* __restrict__ wc,
                         const float* __restrict__ bc, float* __restrict__ xflat,
                         h16* __restrict__ xh16) {
  int i = blockIdx.x * blockDim.x + threadIdx.x;
  if (i >= NB * NDI * NL) return;
  int p = i & 4095; int d = (i >> 12) & 255; int b = i >> 20;
  int h = p >> 6, w = p & 63;
  const float* src = xin + ((size_t)b * NDI + d) * NL;
  const float* wk = wc + d * 9;
  float s = bc[d];
#pragma unroll
  for (int dh = -1; dh <= 1; ++dh)
#pragma unroll
    for (int dw = -1; dw <= 1; ++dw) {
      int hh = h + dh, ww = w + dw;
      if (hh >= 0 && hh < NH && ww >= 0 && ww < NW)
        s += wk[(dh + 1) * 3 + (dw + 1)] * src[hh * NW + ww];
    }
  s = s * dsigmoid(s);                    // SiLU
  xflat[i] = s;                           // (B,DI,L)
  xh16[((size_t)b * NL + p) * NDI + d] = static_cast<h16>(s);  // (B,L,DI) for WMMA A
}

// ---------------- GEMM 3: x_proj per (b,k) with scan-order row remap ----------------
__global__ void k_gemm_xproj(const h16* __restrict__ Xh, const h16* __restrict__ Bw,
                             h16* __restrict__ xdbl16, float* __restrict__ xdbl32) {
  int lane = threadIdx.x & 31;
  int tile = blockIdx.x * 8 + (threadIdx.x >> 5);
  const int NT = 3, SM = NL / (16 * MREP);
  if (tile >= SM * NT) return;
  int bk = blockIdx.y; int b = bk >> 2, k = bk & 3;
  int tm = (tile / NT) * (16 * MREP), tn = (tile % NT) * 16;
  v8f acc[MREP];
  wmma_super<NDI, MREP>(
      [&](int mt) {
        int p = scan_map(k, tm + mt * 16 + (lane & 15));
        return Xh + ((size_t)b * NL + p) * NDI;
      },
      Bw + ((size_t)k * 48 + tn + (lane & 15)) * NDI, lane, acc);
  int n = tn + (lane & 15);
#pragma unroll
  for (int mt = 0; mt < MREP; ++mt)
#pragma unroll
    for (int r = 0; r < 8; ++r) {
      int lo = tm + mt * 16 + r + ((lane >> 4) << 3);
      size_t o = ((size_t)bk * NL + lo) * 48 + n;
      xdbl16[o] = static_cast<h16>(acc[mt][r]);
      xdbl32[o] = acc[mt][r];
    }
}

// ---------------- GEMM 4: dt_proj (K padded to 32) + bias + softplus -> delta ----------------
__global__ void k_gemm_dtproj(const h16* __restrict__ Xd, const h16* __restrict__ Bw,
                              const float* __restrict__ dtb, float* __restrict__ delta) {
  int lane = threadIdx.x & 31;
  int tile = blockIdx.x * 8 + (threadIdx.x >> 5);
  const int NT = NDI / 16, SM = NL / (16 * MREP);
  if (tile >= SM * NT) return;
  int bk = blockIdx.y; int k = bk & 3;
  int tm = (tile / NT) * (16 * MREP), tn = (tile % NT) * 16;
  v8f acc[MREP];
  wmma_super<32, MREP>(
      [&](int mt) { return Xd + ((size_t)bk * NL + tm + mt * 16 + (lane & 15)) * 48; },
      Bw + ((size_t)k * NDI + tn + (lane & 15)) * 32, lane, acc);
  int n = tn + (lane & 15);
  float bias = dtb[k * NDI + n];
#pragma unroll
  for (int mt = 0; mt < MREP; ++mt)
#pragma unroll
    for (int r = 0; r < 8; ++r) {
      int l = tm + mt * 16 + r + ((lane >> 4) << 3);
      delta[((size_t)bk * NDI + n) * NL + l] = dsoftplus(acc[mt][r] + bias);
    }
}

// ---------------- selective scan: one lane per (b,k,d), state N=16 in registers ----------------
__global__ void k_scan(const float* __restrict__ delta, const float* __restrict__ xflat,
                       const float* __restrict__ xdbl32, const float* __restrict__ A_logs,
                       const float* __restrict__ Ds, float* __restrict__ y_pre) {
  int t = blockIdx.x * blockDim.x + threadIdx.x;
  if (t >= NB * NK * NDI) return;
  int d = t & 255; int k = (t >> 8) & 3; int b = t >> 10;
  int kd = k * NDI + d;
  float Af[NST], h[NST];
#pragma unroll
  for (int n = 0; n < NST; ++n) { Af[n] = -__expf(A_logs[kd * NST + n]); h[n] = 0.0f; }
  float Dv = Ds[kd];
  const float* dl = delta + ((size_t)(b * NK + k) * NDI + d) * NL;
  const float* xd = xdbl32 + (size_t)(b * NK + k) * NL * 48;
  const float* xf = xflat + ((size_t)b * NDI + d) * NL;
  const int PF = 8;
  for (int l = 0; l < NL; ++l) {
    if (l + PF < NL) {  // hide HBM/L2 latency behind the serial exp chain
      __builtin_prefetch((const void*)(xd + (size_t)(l + PF) * 48), 0, 1);
      __builtin_prefetch((const void*)(dl + l + PF), 0, 1);
      __builtin_prefetch((const void*)(xf + scan_map(k, l + PF)), 0, 1);
    }
    int p = scan_map(k, l);
    float dt = dl[l];
    float u = xf[p];
    float dtu = dt * u;
    const float* bc = xd + (size_t)l * 48;
    const float4* bv = reinterpret_cast<const float4*>(bc + 8);   // B_t, 16B-aligned
    const float4* cv = reinterpret_cast<const float4*>(bc + 24);  // C_t
    float y = 0.0f;
#pragma unroll
    for (int q = 0; q < 4; ++q) {
      float4 Bt = bv[q], Ct = cv[q];
      int n0 = q * 4;
      h[n0 + 0] = __expf(dt * Af[n0 + 0]) * h[n0 + 0] + dtu * Bt.x; y += h[n0 + 0] * Ct.x;
      h[n0 + 1] = __expf(dt * Af[n0 + 1]) * h[n0 + 1] + dtu * Bt.y; y += h[n0 + 1] * Ct.y;
      h[n0 + 2] = __expf(dt * Af[n0 + 2]) * h[n0 + 2] + dtu * Bt.z; y += h[n0 + 2] * Ct.z;
      h[n0 + 3] = __expf(dt * Af[n0 + 3]) * h[n0 + 3] + dtu * Bt.w; y += h[n0 + 3] * Ct.w;
    }
    y += Dv * u;
    atomicAdd(&y_pre[((size_t)b * NL + p) * NDI + d], y);  // merge 4 directions in place
  }
}

// ---------------- LayerNorm over DI + SiLU(z) gate -> f16 A for out_proj ----------------
__global__ void k_ln_gate(const float* __restrict__ y_pre, const float* __restrict__ zbuf,
                          const float* __restrict__ g, const float* __restrict__ bt,
                          h16* __restrict__ out) {
  int r = blockIdx.x, d = threadIdx.x;
  __shared__ float s1[NDI], s2[NDI];
  float v = y_pre[(size_t)r * NDI + d];
  s1[d] = v; s2[d] = v * v;
  __syncthreads();
  for (int off = 128; off > 0; off >>= 1) {
    if (d < off) { s1[d] += s1[d + off]; s2[d] += s2[d + off]; }
    __syncthreads();
  }
  float mu = s1[0] * (1.0f / NDI);
  float var = s2[0] * (1.0f / NDI) - mu * mu;
  float yv = (v - mu) * rsqrtf(var + 1e-5f) * g[d] + bt[d];
  float z = zbuf[(size_t)r * NDI + d];
  yv *= z * dsigmoid(z);
  out[(size_t)r * NDI + d] = static_cast<h16>(yv);
}

// ---------------- GEMM 5: out_proj, fused mask attention ----------------
__global__ void k_gemm_outproj(const h16* __restrict__ A, const h16* __restrict__ Bw,
                               const float* __restrict__ mask, const float* __restrict__ mg,
                               const float* __restrict__ mb, const float* __restrict__ mm,
                               const float* __restrict__ mv, h16* __restrict__ attended) {
  int lane = threadIdx.x & 31;
  int tile = blockIdx.x * 8 + (threadIdx.x >> 5);
  const int NT = NC / 16, SM = NB * NL / (16 * MREP);
  if (tile >= SM * NT) return;
  int tm = (tile / NT) * (16 * MREP), tn = (tile % NT) * 16;
  v8f acc[MREP];
  wmma_super<NDI, MREP>(
      [&](int mt) { return A + (size_t)(tm + mt * 16 + (lane & 15)) * NDI; },
      Bw + (size_t)(tn + (lane & 15)) * NDI, lane, acc);
  int n = tn + (lane & 15);
  float mgv = mg[0], mbv = mb[0], mmv = mm[0], mrs = rsqrtf(mv[0] + 1e-5f);
#pragma unroll
  for (int mt = 0; mt < MREP; ++mt)
#pragma unroll
    for (int r = 0; r < 8; ++r) {
      int m = tm + mt * 16 + r + ((lane >> 4) << 3);
      int b = m >> 12, p = m & 4095;
      float mn = (mask[(size_t)b * NL + p] - mmv) * mrs * mgv + mbv;
      float sc = 1.0f - dsigmoid(mn);
      attended[(size_t)m * NC + n] = static_cast<h16>(acc[mt][r] * sc);
    }
}

// ---------------- GEMM 6: post_w + BN + GELU + gated residual -> d_out (B,C,H,W) ----------------
__global__ void k_gemm_post(const h16* __restrict__ A, const h16* __restrict__ Bw,
                            const float* __restrict__ g, const float* __restrict__ bt,
                            const float* __restrict__ mn, const float* __restrict__ vr,
                            const float* __restrict__ feat, const float* __restrict__ gate,
                            float* __restrict__ outp) {
  int lane = threadIdx.x & 31;
  int tile = blockIdx.x * 8 + (threadIdx.x >> 5);
  const int NT = NC / 16, SM = NB * NL / (16 * MREP);
  if (tile >= SM * NT) return;
  int tm = (tile / NT) * (16 * MREP), tn = (tile % NT) * 16;
  v8f acc[MREP];
  wmma_super<NC, MREP>(
      [&](int mt) { return A + (size_t)(tm + mt * 16 + (lane & 15)) * NC; },
      Bw + (size_t)(tn + (lane & 15)) * NC, lane, acc);
  int n = tn + (lane & 15);
  float gg = g[n], bb = bt[n], mmv = mn[n], rs = rsqrtf(vr[n] + 1e-5f);
  float gt = gate[0];
#pragma unroll
  for (int mt = 0; mt < MREP; ++mt)
#pragma unroll
    for (int r = 0; r < 8; ++r) {
      int m = tm + mt * 16 + r + ((lane >> 4) << 3);
      int b = m >> 12, p = m & 4095;
      float ref = dgelu((acc[mt][r] - mmv) * rs * gg + bb);
      size_t o = ((size_t)b * NC + n) * NL + p;
      outp[o] = feat[o] + gt * ref;
    }
}

extern "C" void kernel_launch(void* const* d_in, const int* in_sizes, int n_in,
                              void* d_out, int out_size, void* d_ws, size_t ws_size,
                              hipStream_t stream) {
  (void)in_sizes; (void)n_in; (void)out_size; (void)ws_size;
  const float* feature   = (const float*)d_in[0];
  const float* mask_pred = (const float*)d_in[1];
  const float* bn1_g = (const float*)d_in[2];
  const float* bn1_b = (const float*)d_in[3];
  const float* bn1_m = (const float*)d_in[4];
  const float* bn1_v = (const float*)d_in[5];
  const float* pre_w = (const float*)d_in[6];
  const float* pre_g = (const float*)d_in[7];
  const float* pre_b = (const float*)d_in[8];
  const float* pre_m = (const float*)d_in[9];
  const float* pre_v = (const float*)d_in[10];
  const float* in_proj_w = (const float*)d_in[11];
  const float* conv_w = (const float*)d_in[12];
  const float* conv_b = (const float*)d_in[13];
  const float* x_proj_w = (const float*)d_in[14];
  const float* dt_proj_w = (const float*)d_in[15];
  const float* dt_proj_b = (const float*)d_in[16];
  const float* A_logs = (const float*)d_in[17];
  const float* Ds = (const float*)d_in[18];
  const float* out_ln_g = (const float*)d_in[19];
  const float* out_ln_b = (const float*)d_in[20];
  const float* out_proj_w = (const float*)d_in[21];
  const float* mbn_g = (const float*)d_in[22];
  const float* mbn_b = (const float*)d_in[23];
  const float* mbn_m = (const float*)d_in[24];
  const float* mbn_v = (const float*)d_in[25];
  const float* post_w = (const float*)d_in[26];
  const float* post_g = (const float*)d_in[27];
  const float* post_b = (const float*)d_in[28];
  const float* post_m = (const float*)d_in[29];
  const float* post_v = (const float*)d_in[30];
  const float* gate = (const float*)d_in[31];

  char* ws = (char*)d_ws;
  size_t off = 0;
  auto alloc = [&](size_t bytes) -> void* {
    void* p = ws + off;
    off += (bytes + 255) & ~(size_t)255;
    return p;
  };
  h16*   Wpre16   = (h16*)  alloc((size_t)NC * NC * 2);
  h16*   Win16    = (h16*)  alloc((size_t)2 * NDI * NC * 2);
  h16*   Wxp16    = (h16*)  alloc((size_t)NK * 48 * NDI * 2);
  h16*   Wdt16    = (h16*)  alloc((size_t)NK * NDI * 32 * 2);
  h16*   Wout16   = (h16*)  alloc((size_t)NC * NDI * 2);
  h16*   Wpost16  = (h16*)  alloc((size_t)NC * NC * 2);
  h16*   A0       = (h16*)  alloc((size_t)NB * NL * NC * 2);
  h16*   Afeat    = (h16*)  alloc((size_t)NB * NL * NC * 2);
  float* xconv    = (float*)alloc((size_t)NB * NDI * NL * 4);
  float* zbuf     = (float*)alloc((size_t)NB * NL * NDI * 4);
  float* xflat    = (float*)alloc((size_t)NB * NDI * NL * 4);
  h16*   xh16     = (h16*)  alloc((size_t)NB * NL * NDI * 2);
  h16*   xdbl16   = (h16*)  alloc((size_t)NB * NK * NL * 48 * 2);
  float* xdbl32   = (float*)alloc((size_t)NB * NK * NL * 48 * 4);
  float* delta    = (float*)alloc((size_t)NB * NK * NDI * NL * 4);
  float* y_pre    = (float*)alloc((size_t)NB * NL * NDI * 4);
  h16*   ygated16 = (h16*)  alloc((size_t)NB * NL * NDI * 2);
  h16*   attend16 = (h16*)  alloc((size_t)NB * NL * NC * 2);

  const int T = 256;
  // weight packing (cheap, deterministic every call)
  k_cvt_f16<<<(NC * NC + T - 1) / T, T, 0, stream>>>(pre_w, Wpre16, NC * NC);
  k_cvt_f16<<<(2 * NDI * NC + T - 1) / T, T, 0, stream>>>(in_proj_w, Win16, 2 * NDI * NC);
  k_cvt_f16<<<(NC * NDI + T - 1) / T, T, 0, stream>>>(out_proj_w, Wout16, NC * NDI);
  k_cvt_f16<<<(NC * NC + T - 1) / T, T, 0, stream>>>(post_w, Wpost16, NC * NC);
  k_pack_xproj<<<(NK * 48 * NDI + T - 1) / T, T, 0, stream>>>(x_proj_w, Wxp16);
  k_pack_dt<<<(NK * NDI * 32 + T - 1) / T, T, 0, stream>>>(dt_proj_w, Wdt16);

  // pipeline
  k_bn1_pack<<<(NB * NC * NL + T - 1) / T, T, 0, stream>>>(feature, bn1_g, bn1_b, bn1_m, bn1_v, A0);
  k_gemm_pre<<<(NB * NL / (16 * MREP)) * (NC / 16) / 8, T, 0, stream>>>(A0, Wpre16, pre_g, pre_b,
                                                                        pre_m, pre_v, Afeat);
  k_gemm_inproj<<<(NB * NL / (16 * MREP)) * (512 / 16) / 8, T, 0, stream>>>(Afeat, Win16, xconv, zbuf);
  k_dwconv<<<(NB * NDI * NL + T - 1) / T, T, 0, stream>>>(xconv, conv_w, conv_b, xflat, xh16);
  {
    dim3 grid((NL / (16 * MREP)) * 3 / 8, NB * NK);
    k_gemm_xproj<<<grid, T, 0, stream>>>(xh16, Wxp16, xdbl16, xdbl32);
  }
  {
    dim3 grid((NL / (16 * MREP)) * (NDI / 16) / 8, NB * NK);
    k_gemm_dtproj<<<grid, T, 0, stream>>>(xdbl16, Wdt16, dt_proj_b, delta);
  }
  hipMemsetAsync(y_pre, 0, (size_t)NB * NL * NDI * 4, stream);
  k_scan<<<(NB * NK * NDI + T - 1) / T, T, 0, stream>>>(delta, xflat, xdbl32, A_logs, Ds, y_pre);
  k_ln_gate<<<NB * NL, NDI, 0, stream>>>(y_pre, zbuf, out_ln_g, out_ln_b, ygated16);
  k_gemm_outproj<<<(NB * NL / (16 * MREP)) * (NC / 16) / 8, T, 0, stream>>>(ygated16, Wout16, mask_pred,
                                                                            mbn_g, mbn_b, mbn_m, mbn_v,
                                                                            attend16);
  k_gemm_post<<<(NB * NL / (16 * MREP)) * (NC / 16) / 8, T, 0, stream>>>(attend16, Wpost16, post_g,
                                                                         post_b, post_m, post_v,
                                                                         feature, gate, (float*)d_out);
}
